// GAT_LSTM_Model_18176301597354
// MI455X (gfx1250) — matrix operand order repaired and checked by reading, not compiled, not run on hardware
//
#include <hip/hip_runtime.h>
#include <hip/hip_bf16.h>
#include <math.h>

// ---------------------------------------------------------------------------
// Types for CDNA5 WMMA (wave32): 16x16x32 bf16 -> f32 accumulate
// ---------------------------------------------------------------------------
typedef __attribute__((ext_vector_type(16))) __bf16 v16bf;
typedef __attribute__((ext_vector_type(8)))  float  v8f;

#define LEAKY_SLOPE 0.2f

// Monotone float <-> uint encoding so atomicMax(uint) implements float max.
__device__ __forceinline__ unsigned fenc(float f) {
    unsigned u = __float_as_uint(f);
    return (u & 0x80000000u) ? ~u : (u | 0x80000000u);
}
__device__ __forceinline__ float fdec(unsigned u) {
    unsigned b = (u & 0x80000000u) ? (u & 0x7FFFFFFFu) : ~u;
    return __uint_as_float(b);
}

// ---------------------------------------------------------------------------
// Fill kernels (grid-stride)
// ---------------------------------------------------------------------------
__global__ void k_fill_f32(float* p, float v, long n) {
    long i = blockIdx.x * (long)blockDim.x + threadIdx.x;
    long st = (long)gridDim.x * blockDim.x;
    for (; i < n; i += st) p[i] = v;
}
__global__ void k_fill_u32(unsigned* p, unsigned v, long n) {
    long i = blockIdx.x * (long)blockDim.x + threadIdx.x;
    long st = (long)gridDim.x * blockDim.x;
    for (; i < n; i += st) p[i] = v;
}

// ---------------------------------------------------------------------------
// WMMA GEMM: C[M,Ncols] = A[M,K] @ B (+ bias0 (+ bias1))
//   A row-major f32, leading dim LDA (compile-time).
//   TRANSB=0: B row-major [K,Ncols], LDB = Ncols;  B[k,n] = B[k*LDB+n]
//   TRANSB=1: B row-major [Ncols,K], LDB = K;      B[k,n] = B[n*LDB+k]
// All shape parameters except M are compile-time: immediate-offset address
// math, fully unrolled K loop, no 64-bit multiplies in the hot path.
// One wave (32 threads) computes one 16x16 C tile.
// f32 inputs are converted to bf16 in-register; accumulation stays f32.
// VGPR layouts per CDNA5 ISA 7.12.2; EXEC is uniform around the WMMA.
// Interior tiles (row0+16 <= M) take an unguarded store path; only the last
// M-tile uses per-row masking.
// ---------------------------------------------------------------------------
template <int TRANSB, int NBIAS, int LDA, int LDB, int LDC, int K>
__global__ void k_gemm_wmma_bf16(const float* __restrict__ A,
                                 const float* __restrict__ B,
                                 const float* __restrict__ bias0,
                                 const float* __restrict__ bias1,
                                 float* __restrict__ C,
                                 int M)
{
    const int lane = threadIdx.x & 31;
    const int half = lane >> 4;      // 0: lanes 0-15, 1: lanes 16-31
    const int l15  = lane & 15;
    const int row0 = blockIdx.x * 16;
    const int col0 = blockIdx.y * 16;

    int ra = row0 + l15;             // A row held by this lane
    if (ra >= M) ra = M - 1;         // clamp (stores masked below)
    const int nb = col0 + l15;       // B/C column held by this lane

    const float* __restrict__ Arow  = A + (long)ra * LDA;
    const float* __restrict__ Bbase = TRANSB ? (B + (long)nb * LDB) : (B + nb);

    v8f acc = {};
#pragma unroll
    for (int kb = 0; kb < K; kb += 32) {
        v16bf a, b;
        // A 16x32 tile: lanes 0-15 hold K {0-7,16-23}, lanes 16-31 {8-15,24-31}
#pragma unroll
        for (int e = 0; e < 16; ++e) {
            int i = e >> 1;          // VGPR index 0..7
            int k = kb + ((i < 4) ? 0 : 16) + half * 8 + ((i & 3) << 1) + (e & 1);
            a[e] = (__bf16)Arow[k];
        }
        // B 32x16 tile: lane = column; lanes 0-15 hold K {0-7,16-23}, lanes 16-31 {8-15,24-31}
#pragma unroll
        for (int e = 0; e < 16; ++e) {
            int k = kb + ((e < 8) ? 0 : 16) + half * 8 + (e & 7);
            float v = TRANSB ? Bbase[k] : Bbase[(long)k * LDB];
            b[e] = (__bf16)v;
        }
        acc = __builtin_amdgcn_wmma_f32_16x16x32_bf16(
                  /*neg_a=*/false, a, /*neg_b=*/false, b,
                  /*c_mod=*/(short)0, acc, /*reuse_a=*/false, /*reuse_b=*/false);
    }

    float badd = 0.f;
    if (NBIAS >= 1) badd += bias0[nb];
    if (NBIAS >= 2) badd += bias1[nb];

    // C/D layout: VGPR j -> row row0 + 8*half + j, column nb
    float* __restrict__ Cp = C + (long)(row0 + half * 8) * LDC + nb;
    if (row0 + 16 <= M) {            // uniform: interior tile, no masking
#pragma unroll
        for (int j = 0; j < 8; ++j) {
            float o = acc[j];
            if (NBIAS) o += badd;
            Cp[j * LDC] = o;
        }
    } else {                         // boundary tile only
#pragma unroll
        for (int j = 0; j < 8; ++j) {
            int r = row0 + half * 8 + j;
            float o = acc[j];
            if (NBIAS) o += badd;
            if (r < M) Cp[j * LDC] = o;
        }
    }
}

// ---------------------------------------------------------------------------
// GAT pieces
// ---------------------------------------------------------------------------
// per-node attention scores: s[n,h] = sum_c xw[n,h,c]*a_src[h,c]; same for d
__global__ void k_node_scores(const float* __restrict__ xw,
                              const float* __restrict__ a_src,
                              const float* __restrict__ a_dst,
                              float* __restrict__ s, float* __restrict__ d,
                              int N, int H, int C)
{
    int t = blockIdx.x * blockDim.x + threadIdx.x;
    if (t >= N * H) return;
    int n = t / H, h = t % H;
    const float* row = xw + (long)n * H * C + (long)h * C;
    const float* as  = a_src + h * C;
    const float* ad  = a_dst + h * C;
    float ss = 0.f, dd = 0.f;
    for (int c = 0; c < C; ++c) { float v = row[c]; ss += v * as[c]; dd += v * ad[c]; }
    s[t] = ss; d[t] = dd;
}

__device__ __forceinline__ void edge_nodes(long e, long E, const int* srcp, const int* dstp,
                                           int& src, int& dst)
{
    if (e < E) { src = srcp[e]; dst = dstp[e]; }
    else       { src = dst = (int)(e - E); }   // self loops appended
}

// pass 1: segment max over incoming edges (stable softmax)
__global__ void k_edge_max(const int* __restrict__ srcp, const int* __restrict__ dstp,
                           const float* __restrict__ s, const float* __restrict__ d,
                           unsigned* __restrict__ m, long E, int N, int H)
{
    long t = blockIdx.x * (long)blockDim.x + threadIdx.x;
    long tot = (E + N) * (long)H;
    if (t >= tot) return;
    long e = t / H; int h = (int)(t % H);
    int src, dst; edge_nodes(e, E, srcp, dstp, src, dst);
    float v = s[(long)src * H + h] + d[(long)dst * H + h];
    v = (v > 0.f) ? v : LEAKY_SLOPE * v;
    atomicMax(&m[(long)dst * H + h], fenc(v));
}

// pass 2: ex = exp(e - max[dst]); denom[dst] += ex
__global__ void k_edge_exp(const int* __restrict__ srcp, const int* __restrict__ dstp,
                           const float* __restrict__ s, const float* __restrict__ d,
                           const unsigned* __restrict__ m,
                           float* __restrict__ ex, float* __restrict__ den,
                           long E, int N, int H)
{
    long t = blockIdx.x * (long)blockDim.x + threadIdx.x;
    long tot = (E + N) * (long)H;
    if (t >= tot) return;
    long e = t / H; int h = (int)(t % H);
    int src, dst; edge_nodes(e, E, srcp, dstp, src, dst);
    float v = s[(long)src * H + h] + d[(long)dst * H + h];
    v = (v > 0.f) ? v : LEAKY_SLOPE * v;
    float xv = expf(v - fdec(m[(long)dst * H + h]));
    ex[t] = xv;
    atomicAdd(&den[(long)dst * H + h], xv);
}

// pass 3: agg[dst, :] += alpha * xw[src, :] ; one block per edge, thread = channel
__global__ void k_edge_scatter(const int* __restrict__ srcp, const int* __restrict__ dstp,
                               const float* __restrict__ ex, const float* __restrict__ den,
                               const float* __restrict__ xw, float* __restrict__ agg,
                               long E, int N, int H, int C)
{
    long e = blockIdx.x;
    int c = threadIdx.x;             // 0 .. H*C-1
    int h = c / C;
    int src, dst; edge_nodes(e, E, srcp, dstp, src, dst);
    float alpha = ex[e * H + h] / (den[(long)dst * H + h] + 1e-16f);
    atomicAdd(&agg[(long)dst * H * C + c], alpha * xw[(long)src * H * C + c]);
}

// out = act(out + bias[c]); act: 0 = none, 1 = ELU(alpha=1)
__global__ void k_bias_act(float* __restrict__ p, const float* __restrict__ bias,
                           long N, int C, int act)
{
    long t = blockIdx.x * (long)blockDim.x + threadIdx.x;
    if (t >= N * (long)C) return;
    int c = (int)(t % C);
    float v = p[t] + bias[c];
    if (act) v = (v > 0.f) ? v : (expf(v) - 1.f);
    p[t] = v;
}

// ---------------------------------------------------------------------------
// LSTM cell with zero initial state: gates = [i f g o] (torch order),
// c = sigmoid(i)*tanh(g); h = sigmoid(o)*tanh(c). (f*c0 == 0)
// ---------------------------------------------------------------------------
__global__ void k_lstm_zero(const float* __restrict__ gates,
                            float* __restrict__ h, float* __restrict__ c,
                            int N, int HC)
{
    int t = blockIdx.x * blockDim.x + threadIdx.x;
    if (t >= N * HC) return;
    int n = t / HC, j = t % HC;
    const float* g4 = gates + (long)n * 4 * HC;
    float gi = g4[j];
    float gg = g4[2 * HC + j];
    float go = g4[3 * HC + j];
    float cc = (1.f / (1.f + expf(-gi))) * tanhf(gg);
    h[t] = (1.f / (1.f + expf(-go))) * tanhf(cc);
    c[t] = cc;
}

// pred[n] = h1[n,:] . Wo[0,:] + bo
__global__ void k_proj(const float* __restrict__ h1, const float* __restrict__ Wo,
                       const float* __restrict__ bo, float* __restrict__ pred,
                       int N, int HC)
{
    int n = blockIdx.x * blockDim.x + threadIdx.x;
    if (n >= N) return;
    float acc = bo[0];
    for (int j = 0; j < HC; ++j) acc += h1[(long)n * HC + j] * Wo[j];
    pred[n] = acc;
}

// ---------------------------------------------------------------------------
// Host-side launcher
// ---------------------------------------------------------------------------
extern "C" void kernel_launch(void* const* d_in, const int* in_sizes, int n_in,
                              void* d_out, int out_size, void* d_ws, size_t ws_size,
                              hipStream_t stream)
{
    (void)n_in; (void)out_size; (void)ws_size;

    const float* x    = (const float*)d_in[0];
    const int*   ei   = (const int*)d_in[1];   // [2,E] int32
    const float* W1   = (const float*)d_in[3]; // [64,128]
    const float* a1s  = (const float*)d_in[4]; // [4,32]
    const float* a1d  = (const float*)d_in[5];
    const float* b1   = (const float*)d_in[6]; // [128]
    const float* W2   = (const float*)d_in[7]; // [128,64]
    const float* a2s  = (const float*)d_in[8]; // [1,64]
    const float* a2d  = (const float*)d_in[9];
    const float* b2   = (const float*)d_in[10];// [64]
    const float* Wih0 = (const float*)d_in[11];// [256,64]
    const float* bih0 = (const float*)d_in[13];
    const float* bhh0 = (const float*)d_in[14];
    const float* Wih1 = (const float*)d_in[15];// [256,64]
    const float* bih1 = (const float*)d_in[17];
    const float* bhh1 = (const float*)d_in[18];
    const float* Wo   = (const float*)d_in[19];// [1,64]
    const float* bo   = (const float*)d_in[20];

    const int  INF_ = 64;                 // in_channels
    const int  N  = in_sizes[0] / INF_;   // 50000
    const long E  = in_sizes[1] / 2;      // 800000
    const long Ep = E + N;                // with self loops
    const int  H1 = 4, C1 = 32, F1 = 128; // layer-1 heads/chan, F1 = H1*C1
    const int  F2 = 64;                   // layer-2 out channels (1 head)
    const int  HC = 64, G4 = 256;         // LSTM hidden, 4*HC

    const int* srcp = ei;
    const int* dstp = ei + E;

    // ---- workspace bump allocator ----
    char* wsb = (char*)d_ws;
    size_t off = 0;
    auto alloc = [&](size_t bytes) -> char* {
        char* p = wsb + off;
        off += (bytes + 255) & ~(size_t)255;
        return p;
    };
    float*    xw1  = (float*)   alloc((size_t)N * F1 * 4);
    float*    s1   = (float*)   alloc((size_t)N * H1 * 4);
    float*    d1   = (float*)   alloc((size_t)N * H1 * 4);
    unsigned* m1   = (unsigned*)alloc((size_t)N * H1 * 4);
    float*    den1 = (float*)   alloc((size_t)N * H1 * 4);
    float*    ex1  = (float*)   alloc((size_t)Ep * H1 * 4);
    float*    agg1 = (float*)   alloc((size_t)N * F1 * 4);  // becomes h after bias+ELU
    float*    xw2  = (float*)   alloc((size_t)N * F2 * 4);
    float*    s2   = (float*)   alloc((size_t)N * 4);
    float*    d2   = (float*)   alloc((size_t)N * 4);
    unsigned* m2   = (unsigned*)alloc((size_t)N * 4);
    float*    den2 = (float*)   alloc((size_t)N * 4);
    float*    ex2  = (float*)   alloc((size_t)Ep * 4);
    float*    agg2 = (float*)   alloc((size_t)N * F2 * 4);  // becomes g after bias
    float*    gates= (float*)   alloc((size_t)N * G4 * 4);  // reused for both cells

    // ---- output layout: pred[N], h0, h1, c0, c1 ----
    float* pred = (float*)d_out;
    float* h0   = pred + N;
    float* h1p  = h0 + (size_t)N * HC;
    float* c0   = h1p + (size_t)N * HC;
    float* c1   = c0 + (size_t)N * HC;

    const unsigned NEG_INF_ENC = 0x007FFFFFu;   // fenc(-inf)
    const dim3 blk(256);
    auto g1 = [](long n) { return dim3((unsigned)((n + 255) / 256)); };
    const int mt = (N + 15) / 16;               // M tiles

    // ===== GAT layer 1: xw1 = x @ W1 =====
    k_gemm_wmma_bf16<0, 0, /*LDA*/64, /*LDB*/128, /*LDC*/128, /*K*/64>
        <<<dim3(mt, F1 / 16), dim3(32), 0, stream>>>(x, W1, nullptr, nullptr, xw1, N);
    k_node_scores<<<g1((long)N * H1), blk, 0, stream>>>(xw1, a1s, a1d, s1, d1, N, H1, C1);
    k_fill_u32<<<g1((long)N * H1), blk, 0, stream>>>(m1, NEG_INF_ENC, (long)N * H1);
    k_fill_f32<<<g1((long)N * H1), blk, 0, stream>>>(den1, 0.f, (long)N * H1);
    k_fill_f32<<<g1((long)N * F1), blk, 0, stream>>>(agg1, 0.f, (long)N * F1);
    k_edge_max    <<<g1(Ep * H1), blk, 0, stream>>>(srcp, dstp, s1, d1, m1, E, N, H1);
    k_edge_exp    <<<g1(Ep * H1), blk, 0, stream>>>(srcp, dstp, s1, d1, m1, ex1, den1, E, N, H1);
    k_edge_scatter<<<dim3((unsigned)Ep), dim3(H1 * C1), 0, stream>>>(
        srcp, dstp, ex1, den1, xw1, agg1, E, N, H1, C1);
    k_bias_act<<<g1((long)N * F1), blk, 0, stream>>>(agg1, b1, N, F1, /*ELU*/1);

    // ===== GAT layer 2: xw2 = h @ W2 (1 head, concat=False -> identity mean) =====
    k_gemm_wmma_bf16<0, 0, /*LDA*/128, /*LDB*/64, /*LDC*/64, /*K*/128>
        <<<dim3(mt, F2 / 16), dim3(32), 0, stream>>>(agg1, W2, nullptr, nullptr, xw2, N);
    k_node_scores<<<g1((long)N), blk, 0, stream>>>(xw2, a2s, a2d, s2, d2, N, 1, F2);
    k_fill_u32<<<g1((long)N), blk, 0, stream>>>(m2, NEG_INF_ENC, (long)N);
    k_fill_f32<<<g1((long)N), blk, 0, stream>>>(den2, 0.f, (long)N);
    k_fill_f32<<<g1((long)N * F2), blk, 0, stream>>>(agg2, 0.f, (long)N * F2);
    k_edge_max    <<<g1(Ep), blk, 0, stream>>>(srcp, dstp, s2, d2, m2, E, N, 1);
    k_edge_exp    <<<g1(Ep), blk, 0, stream>>>(srcp, dstp, s2, d2, m2, ex2, den2, E, N, 1);
    k_edge_scatter<<<dim3((unsigned)Ep), dim3(F2), 0, stream>>>(
        srcp, dstp, ex2, den2, xw2, agg2, E, N, 1, F2);
    k_bias_act<<<g1((long)N * F2), blk, 0, stream>>>(agg2, b2, N, F2, /*none*/0);

    // ===== LSTM cell 0: gates = g @ Wih0^T + bih0 + bhh0 =====
    k_gemm_wmma_bf16<1, 2, /*LDA*/64, /*LDB*/64, /*LDC*/256, /*K*/64>
        <<<dim3(mt, G4 / 16), dim3(32), 0, stream>>>(agg2, Wih0, bih0, bhh0, gates, N);
    k_lstm_zero<<<g1((long)N * HC), blk, 0, stream>>>(gates, h0, c0, N, HC);

    // ===== LSTM cell 1: gates = h0 @ Wih1^T + bih1 + bhh1 =====
    k_gemm_wmma_bf16<1, 2, /*LDA*/64, /*LDB*/64, /*LDC*/256, /*K*/64>
        <<<dim3(mt, G4 / 16), dim3(32), 0, stream>>>(h0, Wih1, bih1, bhh1, gates, N);
    k_lstm_zero<<<g1((long)N * HC), blk, 0, stream>>>(gates, h1p, c1, N, HC);

    // ===== output projection =====
    k_proj<<<g1((long)N), blk, 0, stream>>>(h1p, Wo, bo, pred, N, HC);
}
